// MultiHeadSelfAttention_18648747999546
// MI455X (gfx1250) — matrix operand order repaired and checked by reading, not compiled, not run on hardware
//
#include <hip/hip_runtime.h>
#include <hip/hip_bf16.h>
#include <math.h>

// ---------------------------------------------------------------------------
// Problem constants (from reference)
// ---------------------------------------------------------------------------
#define BATCH 8
#define SEQ   1024
#define HID   768
#define HEADS 12
#define HDIM  64
#define QKVN  (3 * HID)        // 2304
#define MTOT  (BATCH * SEQ)    // 8192

typedef __attribute__((ext_vector_type(8)))  __bf16 v8bf;
typedef __attribute__((ext_vector_type(16))) __bf16 v16bf;
typedef __attribute__((ext_vector_type(8)))  float  v8f;

union ABFrag { v16bf v; v8bf h[2]; };

static __device__ __forceinline__ v8f wmma_bf16(v16bf a, v16bf b, v8f c) {
  // D = A(16x32 bf16) x B(32x16 bf16) + C(16x16 f32)
  return __builtin_amdgcn_wmma_f32_16x16x32_bf16(
      /*neg_a=*/false, a, /*neg_b=*/false, b,
      /*c_mod=*/(short)0, c, /*reuse_a=*/false, /*reuse_b=*/false);
}

// A fragment 16x32 (M x K). src row-major, leading dim ld (elements), base at [row0][k0].
// ISA layout: lane<16 -> row=lane, K elems {0..7,16..23}; lane>=16 -> row=lane-16, K {8..15,24..31}.
static __device__ __forceinline__ v16bf load_a_frag(const __bf16* base, int ld, int lane) {
  const __bf16* p = base + (size_t)(lane & 15) * ld + ((lane >> 4) << 3);
  ABFrag f;
  f.h[0] = *(const v8bf*)(p);        // K kb..kb+7
  f.h[1] = *(const v8bf*)(p + 16);   // K kb+16..kb+23
  return f.v;
}

// B fragment 32x16 (K x N): column n of B is row n of `base` (row-major, ld elems).
// Contraction values contiguous at element offset coff.
// ISA layout: lane -> n = lane&15; elems i<8 -> kk = i + 8*(lane>>4), i>=8 -> kk = 16 + 8*(lane>>4) + (i-8).
static __device__ __forceinline__ v16bf load_b_frag(const __bf16* base, int ld, int coff, int lane) {
  const __bf16* p = base + (size_t)(lane & 15) * ld + coff + ((lane >> 4) << 3);
  ABFrag f;
  f.h[0] = *(const v8bf*)(p);        // kk 8h..8h+7
  f.h[1] = *(const v8bf*)(p + 16);   // kk 16+8h..16+8h+7
  return f.v;
}

// ---------------------------------------------------------------------------
// Kernel 0: fp32 -> bf16 cast (sizes are all multiples of 4)
// ---------------------------------------------------------------------------
__global__ void cvt_f32_bf16(const float* __restrict__ src, __bf16* __restrict__ dst, int n) {
  int i = (blockIdx.x * blockDim.x + threadIdx.x) * 4;
  if (i + 3 < n) {
    const float4 f = *(const float4*)(src + i);
    dst[i + 0] = (__bf16)f.x;
    dst[i + 1] = (__bf16)f.y;
    dst[i + 2] = (__bf16)f.z;
    dst[i + 3] = (__bf16)f.w;
  }
}

// ---------------------------------------------------------------------------
// Kernels 1 & 3: C[M x Ncols] = A[M x K] * W[Ncols x K]^T  (both bf16, k-contig)
// 128 threads = 4 waves in a 2x2 grid; wave tile 32x64 (2x4 WMMA frags);
// block tile 64x128. 6 fragment loads feed 8 WMMAs per 32-deep k-step.
// MODE 0: QKV epilogue -> scatter bf16 q (scaled), k, v^T.
// MODE 1: proj epilogue -> fp32 out + bias.
// ---------------------------------------------------------------------------
template <int MODE>
__global__ __launch_bounds__(128) void gemm_bf16_wmma(
    const __bf16* __restrict__ A, const __bf16* __restrict__ W, int K,
    __bf16* __restrict__ qbuf, __bf16* __restrict__ kbuf, __bf16* __restrict__ vtbuf,
    float* __restrict__ outp, const float* __restrict__ bias) {
  const int lane  = threadIdx.x & 31;
  const int wave  = threadIdx.x >> 5;
  const int mbase = blockIdx.y * 64 + (wave >> 1) * 32;
  const int nbase = blockIdx.x * 128 + (wave & 1) * 64;

  v8f acc[2][4] = {};
  const __bf16* a0p = A + (size_t)mbase * K;
  const __bf16* a1p = A + (size_t)(mbase + 16) * K;

  for (int k0 = 0; k0 < K; k0 += 32) {
    const v16bf a0 = load_a_frag(a0p + k0, K, lane);
    const v16bf a1 = load_a_frag(a1p + k0, K, lane);
#pragma unroll
    for (int ni = 0; ni < 4; ++ni) {
      const v16bf bf = load_b_frag(W + (size_t)(nbase + ni * 16) * K, K, k0, lane);
      acc[0][ni] = wmma_bf16(a0, bf, acc[0][ni]);
      acc[1][ni] = wmma_bf16(a1, bf, acc[1][ni]);
    }
  }

  const int halfsel = lane >> 4;   // C-frag: elem i holds row (i + 8*halfsel), col = lane&15
  const int ncol    = lane & 15;
#pragma unroll
  for (int mi = 0; mi < 2; ++mi) {
#pragma unroll
    for (int ni = 0; ni < 4; ++ni) {
      const v8f c = acc[mi][ni];
      const int o = nbase + ni * 16 + ncol;
      if (MODE == 0) {
        const int which = o / HID;   // 0=q, 1=k, 2=v
        const int r     = o % HID;
        const int hh    = r >> 6;
        const int dd    = r & 63;
#pragma unroll
        for (int i = 0; i < 8; ++i) {
          const int m  = mbase + mi * 16 + i + 8 * halfsel;
          const int bb = m >> 10;
          const int nn = m & (SEQ - 1);
          const int bh = bb * HEADS + hh;
          const float v = c[i];
          if (which == 0) {
            // fold in 1/sqrt(HEAD_DIM) = 0.125 here
            qbuf[((size_t)bh * SEQ + nn) * HDIM + dd] = (__bf16)(v * 0.125f);
          } else if (which == 1) {
            kbuf[((size_t)bh * SEQ + nn) * HDIM + dd] = (__bf16)v;
          } else {
            // v stored transposed: [b,h,d,n] so P^T x V^T A-frags are contiguous loads
            vtbuf[((size_t)bh * HDIM + dd) * SEQ + nn] = (__bf16)v;
          }
        }
      } else {
        const float bo = bias[o];
#pragma unroll
        for (int i = 0; i < 8; ++i) {
          const int m = mbase + mi * 16 + i + 8 * halfsel;
          outp[(size_t)m * HID + o] = c[i] + bo;
        }
      }
    }
  }
}

// ---------------------------------------------------------------------------
// Kernel 2: flash attention, TRANSPOSED formulation.
//   S^T = K x Q^T   (M=keys, N=queries)  -> each lane owns ONE query column,
//                                           so softmax reduction is in-lane
//                                           (15 VALU ops) + one shfl_xor(16).
//   O^T = V^T x P^T (M=dims, N=queries)  -> V^T tiles are contiguous A-frags,
//                                           P^T is ONE shared B-frag built with
//                                           2 ds_store_b128 + 2 ds_load_b128.
// 4 waves per block, 16 queries per wave.
// ---------------------------------------------------------------------------
__global__ __launch_bounds__(128) void attn_wmma(
    const __bf16* __restrict__ qbuf, const __bf16* __restrict__ kbuf,
    const __bf16* __restrict__ vtbuf, __bf16* __restrict__ ao) {
  __shared__ __bf16 plds[4][16 * 32];   // per wave: P^T tile, [query][key] 16x32

  const int lane = threadIdx.x & 31;
  const int wave = threadIdx.x >> 5;
  const int bh   = blockIdx.x;               // b*HEADS + h
  const int q0   = blockIdx.y * 64 + wave * 16;
  const int b    = bh / HEADS;
  const int h    = bh % HEADS;

  const __bf16* qp = qbuf + (size_t)bh * SEQ * HDIM;
  const __bf16* kp = kbuf + (size_t)bh * SEQ * HDIM;
  const __bf16* vp = vtbuf + (size_t)bh * HDIM * SEQ;

  // Q^T B-fragments (column n = query row q0+n of q; loop-invariant).
  // Two 32-deep contraction steps cover HDIM=64. q is pre-scaled by 1/8.
  const v16bf qb0 = load_b_frag(qp + (size_t)q0 * HDIM, HDIM, 0, lane);
  const v16bf qb1 = load_b_frag(qp + (size_t)q0 * HDIM, HDIM, 32, lane);

  v8f ot0{}, ot1{}, ot2{}, ot3{};   // O^T accum: 4 d-tiles x 16 queries
  float m = -INFINITY, l = 0.0f;    // per-lane (per-query) softmax state

  __bf16* pl = plds[wave];
  const int half = lane >> 4;
  const int ncol = lane & 15;       // this lane's query column

  for (int kb = 0; kb < SEQ; kb += 32) {
    // ---- S^T[32 keys x 16 queries] = K x Q^T ----
    v8f st0{}, st1{};
    {
      const v16bf ka00 = load_a_frag(kp + (size_t)kb * HDIM, HDIM, lane);
      const v16bf ka01 = load_a_frag(kp + (size_t)kb * HDIM + 32, HDIM, lane);
      const v16bf ka10 = load_a_frag(kp + (size_t)(kb + 16) * HDIM, HDIM, lane);
      const v16bf ka11 = load_a_frag(kp + (size_t)(kb + 16) * HDIM + 32, HDIM, lane);
      st0 = wmma_bf16(ka00, qb0, st0);
      st0 = wmma_bf16(ka01, qb1, st0);
      st1 = wmma_bf16(ka10, qb0, st1);
      st1 = wmma_bf16(ka11, qb1, st1);
    }

    // ---- online softmax: all 16 scores of this lane's query are IN-LANE ----
    float mb = fmaxf(st0[0], st1[0]);
#pragma unroll
    for (int i = 1; i < 8; ++i) mb = fmaxf(mb, fmaxf(st0[i], st1[i]));
    mb = fmaxf(mb, __shfl_xor(mb, 16, 32));      // merge lane pair (same query)
    const float mnew  = fmaxf(m, mb);
    const float alpha = __expf(m - mnew);

    v8bf pk0, pk1;
    float rs = 0.0f;
#pragma unroll
    for (int i = 0; i < 8; ++i) {
      const float p0 = __expf(st0[i] - mnew);
      const float p1 = __expf(st1[i] - mnew);
      rs += p0 + p1;
      pk0[i] = (__bf16)p0;
      pk1[i] = (__bf16)p1;
    }
    rs += __shfl_xor(rs, 16, 32);                // merge lane pair (same query)
    l = l * alpha + rs;
    m = mnew;

    // rescale O^T by alpha (uniform per lane = per query)
#pragma unroll
    for (int i = 0; i < 8; ++i) {
      ot0[i] *= alpha; ot1[i] *= alpha; ot2[i] *= alpha; ot3[i] *= alpha;
    }

    // ---- P^T -> LDS [query][key]: each lane's 16 probs are key-contiguous ----
    // lane (ncol, half): frag0 keys 8*half..8*half+7, frag1 keys 16+8*half..+7
    __bf16* prow = pl + ncol * 32 + 8 * half;
    *(v8bf*)(prow)      = pk0;
    *(v8bf*)(prow + 16) = pk1;
    // single shared B-frag P^T (32 keys x 16 queries); same-wave DS ops are in-order
    const v16bf pb = load_b_frag(pl, 32, 0, lane);

    // ---- O^T += V^T x P^T (4 d-tiles; V^T A-frags contiguous from [b,h,d,n]) ----
    ot0 = wmma_bf16(load_a_frag(vp + (size_t)(0 * 16) * SEQ + kb, SEQ, lane), pb, ot0);
    ot1 = wmma_bf16(load_a_frag(vp + (size_t)(1 * 16) * SEQ + kb, SEQ, lane), pb, ot1);
    ot2 = wmma_bf16(load_a_frag(vp + (size_t)(2 * 16) * SEQ + kb, SEQ, lane), pb, ot2);
    ot3 = wmma_bf16(load_a_frag(vp + (size_t)(3 * 16) * SEQ + kb, SEQ, lane), pb, ot3);
  }

  // ---- finalize: O^T elem i of tile t = dim (t*16 + 8*half + i), query ncol ----
  // d is element-contiguous -> 4 x 16B stores per lane.
  const float rl = 1.0f / l;
  const int nq = q0 + ncol;
  __bf16* dst = ao + ((size_t)(b * SEQ + nq)) * HID + h * HDIM + 8 * half;
  const v8f* ot[4] = {&ot0, &ot1, &ot2, &ot3};
#pragma unroll
  for (int t = 0; t < 4; ++t) {
    v8bf w;
#pragma unroll
    for (int i = 0; i < 8; ++i) w[i] = (__bf16)((*ot[t])[i] * rl);
    *(v8bf*)(dst + t * 16) = w;
  }
}

// ---------------------------------------------------------------------------
// Host launcher
// ---------------------------------------------------------------------------
extern "C" void kernel_launch(void* const* d_in, const int* in_sizes, int n_in,
                              void* d_out, int out_size, void* d_ws, size_t ws_size,
                              hipStream_t stream) {
  (void)in_sizes; (void)n_in; (void)out_size; (void)ws_size;
  const float* x      = (const float*)d_in[0];
  const float* w_qkv  = (const float*)d_in[1];
  const float* w_proj = (const float*)d_in[2];
  const float* b_proj = (const float*)d_in[3];
  float* out = (float*)d_out;

  char* ws = (char*)d_ws;
  size_t off = 0;
  auto carve = [&](size_t elems) {
    __bf16* p = (__bf16*)(ws + off);
    off = (off + elems * sizeof(__bf16) + 255) & ~(size_t)255;
    return p;
  };
  __bf16* xb  = carve((size_t)MTOT * HID);                  // x in bf16
  __bf16* wqb = carve((size_t)QKVN * HID);                  // w_qkv in bf16
  __bf16* wpb = carve((size_t)HID * HID);                   // w_proj in bf16
  __bf16* qb  = carve((size_t)BATCH * HEADS * SEQ * HDIM);  // q [b,h,n,d] (scaled)
  __bf16* kcb = carve((size_t)BATCH * HEADS * SEQ * HDIM);  // k [b,h,n,d]
  __bf16* vtb = carve((size_t)BATCH * HEADS * HDIM * SEQ);  // v [b,h,d,n]
  __bf16* aob = carve((size_t)MTOT * HID);                  // attn out [b*n, h*d]

  auto cvt = [&](const float* s, __bf16* d, int n) {
    cvt_f32_bf16<<<(n / 4 + 255) / 256, 256, 0, stream>>>(s, d, n);
  };
  cvt(x, xb, MTOT * HID);
  cvt(w_qkv, wqb, QKVN * HID);
  cvt(w_proj, wpb, HID * HID);

  // QKV projection: [8192 x 768] x [2304 x 768]^T  (block tile 64x128)
  gemm_bf16_wmma<0><<<dim3(QKVN / 128, MTOT / 64), 128, 0, stream>>>(
      xb, wqb, HID, qb, kcb, vtb, nullptr, nullptr);

  // Attention: one block per (b*h, 64-query tile)
  attn_wmma<<<dim3(BATCH * HEADS, SEQ / 64), 128, 0, stream>>>(qb, kcb, vtb, aob);

  // Output projection + bias: [8192 x 768] x [768 x 768]^T -> fp32 out
  gemm_bf16_wmma<1><<<dim3(HID / 128, MTOT / 64), 128, 0, stream>>>(
      aob, wpb, HID, nullptr, nullptr, nullptr, out, b_proj);
}
